// LinearAttention_6932077215994
// MI455X (gfx1250) — compile-verified, matrix operand order checked by reading, not run on hardware
//
#include <hip/hip_runtime.h>

// ---------------------------------------------------------------------------
// Delta-rule linear attention forward for MI455X (gfx1250, wave32, WMMA).
// B=1, H=16, S=4096, D=64, CHUNK=64.
//
// Phase 1 (delta_prep, H*Nc=1024 blocks x 256 thr):
//   normalize k, form k_beta/v_beta, S = k_beta @ k^T via v_wmma_f32_16x16x32_f16,
//   column-parallel forward substitution for W = L^{-1} k_beta, U = L^{-1} v_beta,
//   store W,U (f32) to workspace.
// Phase 2 (delta_scan, H=16 blocks x 256 thr): sequential chunk scan per head,
//   kth = W@h and o_inter = q@h via WMMA (f16 operands, f32 acc),
//   out = q*(U - kth) + o_inter, rank-1 state update with clamp/nan_to_num.
// ---------------------------------------------------------------------------

typedef __attribute__((ext_vector_type(16))) _Float16 v16h;
typedef __attribute__((ext_vector_type(8)))  float    v8f;

#define HN 16
#define SN 4096
#define DN 64
#define CN 64
#define NCHUNK (SN / CN)
#define EPSF 1e-6f

// K-pair base column for VGPR v (0..7) within a 16x32 f16 A fragment
// (ISA 7.12.2: VGPR0..3 hold K=0..7 / lanes-hi K=8..15, VGPR4..7 hold K=16..23 / 24..31).
__device__ __forceinline__ int frag_col(int v, int hi) {
    int base = (v < 4) ? (2 * v) : (16 + 2 * (v - 4));
    return base + hi * 8;
}

// A fragment (16x32) from row-major f32 source, converting to f16 on load.
__device__ __forceinline__ v16h frag_a_f32(const float* src, int row, int ld, int k0, int lane) {
    int m = lane & 15, hi = (lane >> 4) & 1;
    v16h a;
#pragma unroll
    for (int v = 0; v < 8; ++v) {
        int c = k0 + frag_col(v, hi);
        a[2 * v]     = (_Float16)src[(row + m) * ld + c];
        a[2 * v + 1] = (_Float16)src[(row + m) * ld + c + 1];
    }
    return a;
}

// A fragment (16x32) from row-major f16 source.
__device__ __forceinline__ v16h frag_a_f16(const _Float16* src, int row, int ld, int k0, int lane) {
    int m = lane & 15, hi = (lane >> 4) & 1;
    v16h a;
#pragma unroll
    for (int v = 0; v < 8; ++v) {
        int c = k0 + frag_col(v, hi);
        a[2 * v]     = src[(row + m) * ld + c];
        a[2 * v + 1] = src[(row + m) * ld + c + 1];
    }
    return a;
}

// B fragment (32x16): B[k][n] = src[k0+k][col+n], row-major f32 source, cvt f16.
__device__ __forceinline__ v16h frag_b_f32(const float* src, int k0, int col, int ld, int lane) {
    int n = lane & 15, hi = (lane >> 4) & 1;
    v16h b;
#pragma unroll
    for (int v = 0; v < 8; ++v) {
        int r = k0 + frag_col(v, hi);
        b[2 * v]     = (_Float16)src[r * ld + col + n];
        b[2 * v + 1] = (_Float16)src[(r + 1) * ld + col + n];
    }
    return b;
}

// ---------------------------------------------------------------------------
// Phase 1
// ---------------------------------------------------------------------------
__global__ __launch_bounds__(256) void delta_prep(const float* __restrict__ kin,
                                                  const float* __restrict__ vin,
                                                  const float* __restrict__ beta,
                                                  float* __restrict__ wsW,
                                                  float* __restrict__ wsU) {
    __shared__ __attribute__((aligned(16))) float    sW[CN * DN];   // raw k -> k_norm*g -> W
    __shared__ __attribute__((aligned(16))) float    sU[CN * DN];   // raw v -> v*g -> U
    __shared__ __attribute__((aligned(16))) float    sS[CN * CN];   // k_beta @ k^T
    __shared__ __attribute__((aligned(16))) _Float16 sKh[CN * DN];  // normalized k (f16)

    const int tid  = threadIdx.x;
    const int lane = tid & 31;
    const int wave = tid >> 5;
    const int head = blockIdx.x / NCHUNK;
    const int chk  = blockIdx.x % NCHUNK;
    const size_t base = ((size_t)head * SN + (size_t)chk * CN) * DN;

    // Stage k -> sW, v -> sU (b128 loads, 16 floats/thread).
    {
        const float4* k4 = (const float4*)(kin + base);
        const float4* v4 = (const float4*)(vin + base);
        float4* w4 = (float4*)sW;
        float4* u4 = (float4*)sU;
#pragma unroll
        for (int t = 0; t < 4; ++t) {
            int idx = tid + t * 256;
            w4[idx] = k4[idx];
            u4[idx] = v4[idx];
        }
    }
    __syncthreads();

    // Row-wise: l2-normalize k, clamp beta, form k_beta (f32, in sW), v_beta, k_norm (f16).
    if (tid < CN) {
        const int r = tid;
        float ss = 0.0f;
#pragma unroll 8
        for (int c = 0; c < DN; ++c) ss += sW[r * DN + c] * sW[r * DN + c];
        const float inv = 1.0f / (sqrtf(ss) + EPSF);
        float g = beta[(size_t)head * SN + (size_t)chk * CN + r];
        g = fminf(fmaxf(g, EPSF), 1.0f - EPSF);
#pragma unroll 8
        for (int c = 0; c < DN; ++c) {
            const float kn = sW[r * DN + c] * inv;
            sKh[r * DN + c] = (_Float16)kn;
            sW[r * DN + c]  = kn * g;   // k_beta
            sU[r * DN + c] *= g;        // v_beta
        }
    }
    __syncthreads();

    // S = k_beta @ k^T : 16 tiles of 16x16, 2 per wave, K=64 in two x32 steps.
#pragma unroll
    for (int tt = 0; tt < 2; ++tt) {
        const int t  = wave * 2 + tt;
        const int ti = t >> 2, tj = t & 3;
        v8f acc = {};
#pragma unroll
        for (int k0 = 0; k0 < DN; k0 += 32) {
            v16h a = frag_a_f32(sW, ti * 16, DN, k0, lane);          // A = k_beta
            v16h b = frag_a_f16(sKh, tj * 16, DN, k0, lane);         // B = k^T (A-style load)
            acc = __builtin_amdgcn_wmma_f32_16x16x32_f16(false, a, false, b,
                                                         (short)0, acc, false, false);
        }
        const int m0 = (lane >> 4) * 8, nn = lane & 15;
#pragma unroll
        for (int r = 0; r < 8; ++r)
            sS[(ti * 16 + m0 + r) * CN + tj * 16 + nn] = acc[r];
    }
    __syncthreads();

    // Unit-lower solve (I + strict_lower(S)) X = rhs, columns independent -> no barriers.
    if (tid < DN) {
        const int c = tid;
        for (int i = 1; i < CN; ++i) {
            float wi = sW[i * DN + c];
            float ui = sU[i * DN + c];
            for (int j = 0; j < i; ++j) {
                const float l = sS[i * CN + j];
                wi = fmaf(-l, sW[j * DN + c], wi);
                ui = fmaf(-l, sU[j * DN + c], ui);
            }
            sW[i * DN + c] = wi;
            sU[i * DN + c] = ui;
        }
    }
    __syncthreads();

    // Store W, U to workspace (f32, b128 stores).
    {
        float4* outW = (float4*)(wsW + base);
        float4* outU = (float4*)(wsU + base);
        const float4* w4 = (const float4*)sW;
        const float4* u4 = (const float4*)sU;
#pragma unroll
        for (int t = 0; t < 4; ++t) {
            int idx = tid + t * 256;
            outW[idx] = w4[idx];
            outU[idx] = u4[idx];
        }
    }
}

// ---------------------------------------------------------------------------
// Phase 2
// ---------------------------------------------------------------------------
__global__ __launch_bounds__(256) void delta_scan(const float* __restrict__ qin,
                                                  const float* __restrict__ wsW,
                                                  const float* __restrict__ wsU,
                                                  float* __restrict__ out) {
    __shared__ __attribute__((aligned(16))) float    sH[DN * DN];   // state h (f32)
    __shared__ __attribute__((aligned(16))) float    sWc[CN * DN];  // W chunk (f32)
    __shared__ __attribute__((aligned(16))) float    sUc[CN * DN];  // q staging -> U -> u_val
    __shared__ __attribute__((aligned(16))) _Float16 sQh[CN * DN];  // normalized q (f16)

    const int tid  = threadIdx.x;
    const int lane = tid & 31;
    const int wave = tid >> 5;
    const int head = blockIdx.x;

#pragma unroll
    for (int t = 0; t < 16; ++t) sH[tid + t * 256] = 0.0f;
    __syncthreads();

    for (int n = 0; n < NCHUNK; ++n) {
        const size_t base = ((size_t)head * SN + (size_t)n * CN) * DN;

        // Stage raw q into sUc (temp).
        {
            const float4* q4 = (const float4*)(qin + base);
            float4* u4 = (float4*)sUc;
#pragma unroll
            for (int t = 0; t < 4; ++t) { int idx = tid + t * 256; u4[idx] = q4[idx]; }
        }
        __syncthreads();

        // Normalize q rows -> sQh (f16).
        if (tid < CN) {
            const int r = tid;
            float ss = 0.0f;
#pragma unroll 8
            for (int c = 0; c < DN; ++c) ss += sUc[r * DN + c] * sUc[r * DN + c];
            const float inv = 1.0f / (sqrtf(ss) + EPSF);
#pragma unroll 8
            for (int c = 0; c < DN; ++c)
                sQh[r * DN + c] = (_Float16)(sUc[r * DN + c] * inv);
        }
        __syncthreads();

        // Stage W, U for this chunk; prefetch next chunk's W/U into L2.
        {
            const float4* w4g = (const float4*)(wsW + base);
            const float4* u4g = (const float4*)(wsU + base);
            float4* w4 = (float4*)sWc;
            float4* u4 = (float4*)sUc;
#pragma unroll
            for (int t = 0; t < 4; ++t) {
                int idx = tid + t * 256;
                w4[idx] = w4g[idx];
                u4[idx] = u4g[idx];
            }
            if (n + 1 < NCHUNK) {
                __builtin_prefetch(wsW + base + (size_t)CN * DN + tid * 16, 0, 1);
                __builtin_prefetch(wsU + base + (size_t)CN * DN + tid * 16, 0, 1);
            }
        }
        __syncthreads();

        // kth = W @ h, o_inter = q @ h ; u_val = U - kth (written back into sUc);
        // out = q * u_val + o_inter.
#pragma unroll
        for (int tt = 0; tt < 2; ++tt) {
            const int t  = wave * 2 + tt;
            const int ti = t >> 2, tj = t & 3;
            v8f ak = {};
            v8f ao = {};
#pragma unroll
            for (int k0 = 0; k0 < DN; k0 += 32) {
                v16h bh = frag_b_f32(sH, k0, tj * 16, DN, lane);     // B = h
                v16h aw = frag_a_f32(sWc, ti * 16, DN, k0, lane);    // A = W
                v16h aq = frag_a_f16(sQh, ti * 16, DN, k0, lane);    // A = q
                ak = __builtin_amdgcn_wmma_f32_16x16x32_f16(false, aw, false, bh,
                                                            (short)0, ak, false, false);
                ao = __builtin_amdgcn_wmma_f32_16x16x32_f16(false, aq, false, bh,
                                                            (short)0, ao, false, false);
            }
            const int m0 = (lane >> 4) * 8, nn = lane & 15;
#pragma unroll
            for (int r = 0; r < 8; ++r) {
                const int row = ti * 16 + m0 + r;
                const int col = tj * 16 + nn;
                const float uv = sUc[row * DN + col] - ak[r];
                sUc[row * DN + col] = uv;                            // u_val (last row feeds state)
                const float qv = (float)sQh[row * DN + col];
                out[base + (size_t)row * DN + col] = fmaf(qv, uv, ao[r]);
            }
        }
        __syncthreads();

        // h += outer(W[63,:], u_val[63,:]); clamp to +-1e4; NaN -> 0.
#pragma unroll
        for (int t = 0; t < 16; ++t) {
            const int idx = tid * 16 + t;
            const int r = idx >> 6, c = idx & 63;
            float hn = fmaf(sWc[63 * DN + r], sUc[63 * DN + c], sH[idx]);
            hn = (hn != hn) ? 0.0f : fminf(fmaxf(hn, -10000.0f), 10000.0f);
            sH[idx] = hn;
        }
        __syncthreads();
    }
}

// ---------------------------------------------------------------------------
// Launch
// ---------------------------------------------------------------------------
extern "C" void kernel_launch(void* const* d_in, const int* in_sizes, int n_in,
                              void* d_out, int out_size, void* d_ws, size_t ws_size,
                              hipStream_t stream) {
    (void)in_sizes; (void)n_in; (void)out_size; (void)ws_size;
    const float* q    = (const float*)d_in[0];
    const float* k    = (const float*)d_in[1];
    const float* v    = (const float*)d_in[2];
    const float* beta = (const float*)d_in[3];
    float* outp = (float*)d_out;

    float* wsW = (float*)d_ws;                          // [H, S, D] f32
    float* wsU = wsW + (size_t)HN * SN * DN;            // [H, S, D] f32

    delta_prep<<<dim3(HN * NCHUNK), dim3(256), 0, stream>>>(k, v, beta, wsW, wsU);
    delta_scan<<<dim3(HN), dim3(256), 0, stream>>>(q, wsW, wsU, outp);
}
